// KANConv2D_31379031065004
// MI455X (gfx1250) — matrix operand order repaired and checked by reading, not compiled, not run on hardware
//
#include <hip/hip_runtime.h>
#include <math.h>

typedef __attribute__((ext_vector_type(16))) _Float16 v16h;
typedef __attribute__((ext_vector_type(8)))  _Float16 v8h;
typedef __attribute__((ext_vector_type(8)))  float    v8f;

#define BDIM   32
#define HDIM   64
#define WDIM   64
#define CDIM   64
#define FDIM   128
#define PDIM   576
#define NPATCH (BDIM*HDIM*WDIM)   // 131072
#define KSTEPS 18                 // 576 / 32
#define BM     64                 // rows per block
#define LDSTR  40                 // f16 per A-row in LDS (32 + 8 pad)

// ---------------------------------------------------------------------------
// ws layout (floats):
//   [0]          gamma
//   [1]          S_pn (atomic sum of all patch norms)
//   [64..640)    aggV[9][64]  per-channel boundary sums of input
//   [1280..1408) cn[128]      per-filter ||c_f||^2
//   [1408..1984) rsum[576]    per-k row-sum of control points
//   [2048..+N)   pn[N]        per-patch ||p||^2
//   [+N..+2N)    rowsq[N]     per-position channel sum of squares
//   then (16B aligned): wk_pk[73728] f16, cp_pk[73728] f16  (WMMA-packed B)
// ---------------------------------------------------------------------------

// Pre-pack weights (576x128 fp32, row-major, k=(i*3+j)*64+c) into WMMA
// B-fragment order for v_wmma_f32_16x16x32_f16:
//   B is 32x16 per fragment; lane l: col = l%16, elements e=0..15 hold
//   K = 16*(l/16) + e (contiguous halves, same order as the A fragments in kan_main).
__global__ void kan_pack_w(const float* __restrict__ wk,
                           const float* __restrict__ cp,
                           _Float16* __restrict__ wk_pk,
                           _Float16* __restrict__ cp_pk) {
  int blk   = blockIdx.x;        // 0..143 = kstep*8 + ftile
  int kstep = blk >> 3;
  int ftile = blk & 7;
  int t     = threadIdx.x;       // 0..255
  int lane  = t >> 3;            // 0..31
  int d     = t & 7;             // dword pair index 0..7
  int half  = lane >> 4;
  int col   = ftile * 16 + (lane & 15);
  int k     = kstep * 32 + 16 * half + 2 * d;
  size_t o  = ((size_t)blk * 32 + lane) * 16 + 2 * d;
  wk_pk[o]     = (_Float16)wk[(size_t)k * FDIM + col];
  wk_pk[o + 1] = (_Float16)wk[(size_t)(k + 1) * FDIM + col];
  cp_pk[o]     = (_Float16)cp[(size_t)k * FDIM + col];
  cp_pk[o + 1] = (_Float16)cp[(size_t)(k + 1) * FDIM + col];
}

// cn[f] = sum_k cp[k,f]^2 ; rsum[k] = sum_f cp[k,f]
__global__ void kan_reduce_cp(const float* __restrict__ cp,
                              float* __restrict__ cn,
                              float* __restrict__ rsum) {
  int t = threadIdx.x;  // 576 threads
  if (t < FDIM) {
    float s = 0.f;
    for (int k = 0; k < PDIM; ++k) { float v = cp[(size_t)k * FDIM + t]; s += v * v; }
    cn[t] = s;
  }
  if (t < PDIM) {
    float s = 0.f;
    for (int f = 0; f < FDIM; ++f) s += cp[(size_t)t * FDIM + f];
    rsum[t] = s;
  }
}

// Per-channel boundary aggregates of the input:
// region 0=all 1=row0 2=row63 3=col0 4=col63 5=r0c0 6=r0c63 7=r63c0 8=r63c63
__global__ void kan_reduce_inp(const float* __restrict__ x,
                               float* __restrict__ aggV) {
  __shared__ float s[9 * CDIM];
  int t = threadIdx.x;
  for (int i = t; i < 9 * CDIM; i += blockDim.x) s[i] = 0.f;
  __syncthreads();
  int n = blockIdx.x * blockDim.x + t;            // position index
  int w = n & 63, h = (n >> 6) & 63;
  const float* base = x + (size_t)n * CDIM;
  bool r0 = (h == 0), r63 = (h == 63), c0 = (w == 0), c63 = (w == 63);
  for (int cc = 0; cc < CDIM; ++cc) {
    int c = (cc + t) & 63;                        // stagger to spread LDS banks
    float v = base[c];
    atomicAdd(&s[0 * CDIM + c], v);
    if (r0)  atomicAdd(&s[1 * CDIM + c], v);
    if (r63) atomicAdd(&s[2 * CDIM + c], v);
    if (c0)  atomicAdd(&s[3 * CDIM + c], v);
    if (c63) atomicAdd(&s[4 * CDIM + c], v);
    if (r0 && c0)   atomicAdd(&s[5 * CDIM + c], v);
    if (r0 && c63)  atomicAdd(&s[6 * CDIM + c], v);
    if (r63 && c0)  atomicAdd(&s[7 * CDIM + c], v);
    if (r63 && c63) atomicAdd(&s[8 * CDIM + c], v);
  }
  __syncthreads();
  for (int i = t; i < 9 * CDIM; i += blockDim.x) atomicAdd(&aggV[i], s[i]);
}

// rowsq[b,h,w] = sum_c x^2
__global__ void kan_rowsq(const float* __restrict__ x, float* __restrict__ rowsq) {
  int n = blockIdx.x * blockDim.x + threadIdx.x;
  const float4* p = (const float4*)(x + (size_t)n * CDIM);
  float s = 0.f;
#pragma unroll
  for (int i = 0; i < 16; ++i) { float4 v = p[i]; s += v.x*v.x + v.y*v.y + v.z*v.z + v.w*v.w; }
  rowsq[n] = s;
}

// pn[b,h,w] = 3x3 window sum of rowsq (zero pad); atomic total into S_pn
__global__ void kan_pn(const float* __restrict__ rowsq,
                       float* __restrict__ pn, float* __restrict__ S_pn) {
  int n = blockIdx.x * blockDim.x + threadIdx.x;
  int w = n & 63, h = (n >> 6) & 63, b = n >> 12;
  float s = 0.f;
  for (int di = -1; di <= 1; ++di)
    for (int dj = -1; dj <= 1; ++dj) {
      int hh = h + di, ww = w + dj;
      if (hh >= 0 && hh < HDIM && ww >= 0 && ww < WDIM)
        s += rowsq[(b << 12) + (hh << 6) + ww];
    }
  pn[n] = s;
  __shared__ float red[256];
  red[threadIdx.x] = s; __syncthreads();
  for (int off = 128; off > 0; off >>= 1) {
    if ((int)threadIdx.x < off) red[threadIdx.x] += red[threadIdx.x + off];
    __syncthreads();
  }
  if (threadIdx.x == 0) atomicAdd(S_pn, red[0]);
}

// gamma = (N*F) / (2 * (F*S_pn + N*S_cn - 2*sum_k colsum_p[k]*rsum[k]))
__global__ void kan_gamma(const float* __restrict__ aggV,
                          const float* __restrict__ cn,
                          const float* __restrict__ rsum,
                          const float* __restrict__ S_pn,
                          float* __restrict__ gamma) {
  __shared__ float redc[64], redn[64];
  int c = threadIdx.x;  // 64 threads, one channel each
  float cross = 0.f;
  for (int i = 0; i < 3; ++i)
    for (int j = 0; j < 3; ++j) {
      float cs = aggV[0 * 64 + c];
      if (i == 0) cs -= aggV[2 * 64 + c];        // i=0 uses rows 0..62
      else if (i == 2) cs -= aggV[1 * 64 + c];   // i=2 uses rows 1..63
      if (j == 0) cs -= aggV[4 * 64 + c];
      else if (j == 2) cs -= aggV[3 * 64 + c];
      if (i == 0 && j == 0) cs += aggV[8 * 64 + c];
      if (i == 0 && j == 2) cs += aggV[7 * 64 + c];
      if (i == 2 && j == 0) cs += aggV[6 * 64 + c];
      if (i == 2 && j == 2) cs += aggV[5 * 64 + c];
      cross += cs * rsum[(i * 3 + j) * 64 + c];
    }
  redc[c] = cross;
  redn[c] = cn[c] + cn[c + 64];
  __syncthreads();
  for (int off = 32; off > 0; off >>= 1) {
    if (c < off) { redc[c] += redc[c + off]; redn[c] += redn[c + off]; }
    __syncthreads();
  }
  if (c == 0) {
    double sum_d = (double)FDIM * (double)S_pn[0]
                 + (double)NPATCH * (double)redn[0]
                 - 2.0 * (double)redc[0];
    double mean = sum_d / ((double)NPATCH * (double)FDIM);
    gamma[0] = (float)(1.0 / (2.0 * mean));
  }
}

// ---------------------------------------------------------------------------
// Main fused kernel: dual-GEMM (conv + RBF dot) with shared A fragments.
// Block = 256 threads (8 waves) -> 64 rows x 128 cols output tile.
// Wave w owns 16 cols; 4 row-tiles x 2 matrices = 8 WMMAs per K-step.
// ---------------------------------------------------------------------------
__global__ __launch_bounds__(256) void kan_main(
    const float* __restrict__ x,
    const _Float16* __restrict__ wk_pk,
    const _Float16* __restrict__ cp_pk,
    const float* __restrict__ bias,
    const float* __restrict__ cn,
    const float* __restrict__ pn,
    const float* __restrict__ gamma_p,
    float* __restrict__ out) {
  __shared__ _Float16 As[BM * LDSTR];
  __shared__ float pns[BM];

  int t = threadIdx.x;
  int m0 = blockIdx.x * BM;
  if (t < BM) pns[t] = pn[m0 + t];

  int wave = t >> 5, lane = t & 31;
  int half = lane >> 4, m = lane & 15;
  float gamma = gamma_p[0];

  v8f accC[4], accD[4];
  v8f vzero = {};
#pragma unroll
  for (int rt = 0; rt < 4; ++rt) { accC[rt] = vzero; accD[rt] = vzero; }

  // A-tile loader mapping: 4 threads per row, 8 floats each
  int rowA = t >> 2;
  int seg  = t & 3;
  int nn = m0 + rowA;
  int pw = nn & 63, ph = (nn >> 6) & 63, pb = nn >> 12;

  for (int kstep = 0; kstep < KSTEPS; ++kstep) {
    int i  = kstep / 6;             // kernel tap row
    int j  = (kstep >> 1) % 3;      // kernel tap col
    int c0 = (kstep & 1) << 5;      // channel half: 0 or 32

    // ---- stage A tile (64 rows x 32 f16) into LDS ----
    int hh = ph + i - 1, ww = pw + j - 1;
    v8h hv = {};
    if (hh >= 0 && hh < HDIM && ww >= 0 && ww < WDIM) {
      const float4* src = (const float4*)(x +
          ((((size_t)pb * HDIM + hh) * WDIM + ww) * CDIM + c0 + seg * 8));
      float4 v0 = src[0], v1 = src[1];
      hv[0] = (_Float16)v0.x; hv[1] = (_Float16)v0.y;
      hv[2] = (_Float16)v0.z; hv[3] = (_Float16)v0.w;
      hv[4] = (_Float16)v1.x; hv[5] = (_Float16)v1.y;
      hv[6] = (_Float16)v1.z; hv[7] = (_Float16)v1.w;
    }
    *(v8h*)(&As[rowA * LDSTR + seg * 8]) = hv;
    __syncthreads();

    // ---- B fragments for this wave's 16-column slab ----
    size_t bo = (((size_t)kstep * 8 + wave) * 32 + lane) * 16;
    v16h bw = *(const v16h*)(wk_pk + bo);
    v16h bc = *(const v16h*)(cp_pk + bo);

#pragma unroll
    for (int rt = 0; rt < 4; ++rt) {
      // A fragment (16x32 f16), this code's k order, not AMD's table order:
      // lane m, half h: elems 0..15 -> K = 16*h + 0..15 (contiguous, matches packed B)
      int abase = (rt * 16 + m) * LDSTR + 16 * half;
      v8h lo = *(const v8h*)(&As[abase]);
      v8h hi = *(const v8h*)(&As[abase + 8]);
      v16h a;
#pragma unroll
      for (int e = 0; e < 8; ++e) { a[e] = lo[e]; a[e + 8] = hi[e]; }
      accC[rt] = __builtin_amdgcn_wmma_f32_16x16x32_f16(
          false, a, false, bw, (short)0, accC[rt], false, false);
      accD[rt] = __builtin_amdgcn_wmma_f32_16x16x32_f16(
          false, a, false, bc, (short)0, accD[rt], false, false);
    }
    __syncthreads();
  }

  // ---- epilogue: d = pn + cn - 2*dot ; out = conv + exp(-g*d) + bias ----
  int col = (wave << 4) + m;
  float cnc = cn[col];
  float bs  = bias[col];
#pragma unroll
  for (int rt = 0; rt < 4; ++rt) {
#pragma unroll
    for (int v = 0; v < 8; ++v) {
      int row = rt * 16 + v + 8 * half;          // C/D layout: M = v + 8*(lane/16)
      float d = pns[row] + cnc - 2.f * accD[rt][v];
      float o = accC[rt][v] + __expf(-gamma * d) + bs;
      out[(size_t)(m0 + row) * FDIM + col] = o;
    }
  }
}

extern "C" void kernel_launch(void* const* d_in, const int* in_sizes, int n_in,
                              void* d_out, int out_size, void* d_ws, size_t ws_size,
                              hipStream_t stream) {
  (void)in_sizes; (void)n_in; (void)out_size; (void)ws_size;
  const float* x    = (const float*)d_in[0];   // inputs [32,64,64,64]
  const float* wk   = (const float*)d_in[1];   // kernel [3,3,64,128]
  const float* bias = (const float*)d_in[2];   // bias [128]
  const float* cp   = (const float*)d_in[3];   // control_points [3,3,64,128]
  float* out = (float*)d_out;
  float* ws  = (float*)d_ws;

  float* gamma = ws + 0;
  float* S_pn  = ws + 1;
  float* aggV  = ws + 64;                      // 9*64
  float* cn    = ws + 1280;                    // 128
  float* rsum  = ws + 1408;                    // 576
  float* pnb   = ws + 2048;                    // 131072
  float* rowsq = ws + 2048 + NPATCH;           // 131072
  _Float16* wk_pk = (_Float16*)(ws + 2048 + 2 * NPATCH);   // 73728 f16 (16B aligned)
  _Float16* cp_pk = wk_pk + KSTEPS * 8 * 32 * 16;

  hipMemsetAsync(ws, 0, 2048 * sizeof(float), stream);
  kan_pack_w   <<<KSTEPS * 8, 256, 0, stream>>>(wk, cp, wk_pk, cp_pk);
  kan_reduce_cp<<<1, PDIM, 0, stream>>>(cp, cn, rsum);
  kan_reduce_inp<<<NPATCH / 256, 256, 0, stream>>>(x, aggV);
  kan_rowsq    <<<NPATCH / 256, 256, 0, stream>>>(x, rowsq);
  kan_pn       <<<NPATCH / 256, 256, 0, stream>>>(rowsq, pnb, S_pn);
  kan_gamma    <<<1, 64, 0, stream>>>(aggV, cn, rsum, S_pn, gamma);
  kan_main     <<<NPATCH / BM, 256, 0, stream>>>(x, wk_pk, cp_pk, bias, cn, pnb, gamma, out);
}
